// Model_22763326669311
// MI455X (gfx1250) — matrix operand order repaired and checked by reading, not compiled, not run on hardware
//
#include <hip/hip_runtime.h>
#include <hip/hip_bf16.h>

// Sizes from the reference
#define N   1024
#define H   256
#define S   32
#define NH  4
#define DH  64

typedef __attribute__((ext_vector_type(16))) _Float16 v16h;
typedef __attribute__((ext_vector_type(8)))  float    v8f;

// ---------------------------------------------------------------------------
// WMMA fragment layouts used here (CDNA5 16x16x32 f16, wave32; contiguous K order, same for A and B):
//   A (16x32):  lane l -> row = l&15; half h -> K = (l<16 ? h : 16+h)
//   B (32x16):  lane l -> col = l&15; half h -> K = (l<16 ? h : 16+h)
//   C/D (16x16 f32): lane l, vgpr v -> row = v + (l<16?0:8), col = l&15
// ---------------------------------------------------------------------------

// rank-1 graph-builder collapse: ua = W1[:, :H]^T w2, ub = W1[:, H:]^T w2, c
__global__ void k_uabc(const float* __restrict__ W1, const float* __restrict__ b1,
                       const float* __restrict__ w2, const float* __restrict__ b2,
                       float* ua, float* ub, float* cc) {
    int t = threadIdx.x;                       // 256 threads
    float sa = 0.f, sb = 0.f;
    for (int k = 0; k < 128; ++k) {
        float w = w2[k];
        sa += W1[k * (2 * H) + t] * w;
        sb += W1[k * (2 * H) + H + t] * w;
    }
    ua[t] = sa; ub[t] = sb;
    if (t == 0) {
        float c = b2[0];
        for (int k = 0; k < 128; ++k) c += w2[k] * b1[k];
        cc[0] = c;
    }
}

// layernorm rows of embed -> e, plus a_i = e_i.ua, b_i = e_i.ub
__global__ void k_ln1(const float* __restrict__ emb, const float* __restrict__ g,
                      const float* __restrict__ b, const float* __restrict__ ua,
                      const float* __restrict__ ub, float* __restrict__ e,
                      float* __restrict__ av, float* __restrict__ bv) {
    int i = blockIdx.x, t = threadIdx.x;       // 256 threads = H
    __shared__ float red[256];
    float x = emb[i * H + t];
    red[t] = x; __syncthreads();
    for (int o = 128; o; o >>= 1) { if (t < o) red[t] += red[t + o]; __syncthreads(); }
    float mu = red[0] * (1.f / H); __syncthreads();
    float d = x - mu;
    red[t] = d * d; __syncthreads();
    for (int o = 128; o; o >>= 1) { if (t < o) red[t] += red[t + o]; __syncthreads(); }
    float rs = rsqrtf(red[0] * (1.f / H) + 1e-5f); __syncthreads();
    float ev = d * rs * g[t] + b[t];
    e[i * H + t] = ev;
    red[t] = ev * ua[t]; __syncthreads();
    for (int o = 128; o; o >>= 1) { if (t < o) red[t] += red[t + o]; __syncthreads(); }
    float ai = red[0]; __syncthreads();
    red[t] = ev * ub[t]; __syncthreads();
    for (int o = 128; o; o >>= 1) { if (t < o) red[t] += red[t + o]; __syncthreads(); }
    if (t == 0) { av[i] = ai; bv[i] = red[0]; }
}

// fused adjacency -> byte mask adjm[i][j] = (fused > 0)
__global__ void k_adj(const float* __restrict__ adj, const float* __restrict__ av,
                      const float* __restrict__ bv, const float* __restrict__ cc,
                      const float* alpha_p, const float* pw_p, const float* pb_p,
                      unsigned char* __restrict__ adjm) {
    int i = blockIdx.x;
    float pw = pw_p[0], pb = pb_p[0], alpha = alpha_p[0], c = cc[0];
    float ai = av[i], bi = bv[i];
    for (int j = threadIdx.x; j < N; j += 256) {
        float pa = pw * adj[i * N + j] + pb;           // proj_adj[i,j]
        float pt = pw * adj[j * N + i] + pb;           // proj_adj[j,i] (emask.T)
        float dyn = 0.f;
        if (pt > 0.f)       dyn = 1.f / (1.f + __expf(-(av[j] + bi + c)));  // sT wins
        else if (pa > 0.f)  dyn = 1.f / (1.f + __expf(-(ai + bv[j] + c))); // sF
        float fused = pa * (1.f - alpha) + dyn * alpha;
        adjm[i * N + j] = fused > 0.f ? (unsigned char)1 : (unsigned char)0;
    }
}

// f32 -> f16 copy of merge_W (done once, not per step-block)
__global__ void k_cvtw(const float* __restrict__ mW, _Float16* __restrict__ mWh) {
    int i = blockIdx.x * 256 + threadIdx.x;
    mWh[i] = (_Float16)mW[i];
}

// Hh[k] = e @ Wh[k]  (1024x256 @ 256x64), WMMA f16 in / f32 out; also f16 copy
__global__ void k_hh(const float* __restrict__ e, const float* __restrict__ Wh,
                     float* __restrict__ Hh, _Float16* __restrict__ Hh_h) {
    int tile = blockIdx.x, k = blockIdx.y;     // 64 tiles x 4 heads, 128 threads
    int t = threadIdx.x, lane = t & 31, w = t >> 5;
    __shared__ _Float16 at[16 * H];
    for (int idx = t; idx < 16 * H; idx += 128)
        at[idx] = (_Float16)e[(tile * 16 + (idx >> 8)) * H + (idx & 255)];
    __syncthreads();
    int arow = lane & 15, aoff = (lane < 16) ? 0 : 8, bko = (lane < 16) ? 0 : 16;
    int col = w * 16 + (lane & 15);
    v8f acc = {};
    for (int kk = 0; kk < 8; ++kk) {
        int kb = kk * 32;
        v16h a, b;
#pragma unroll
        for (int h = 0; h < 16; ++h) { int K = kb + bko + h; a[h] = at[arow * H + K]; }
#pragma unroll
        for (int h = 0; h < 16; ++h) { int K = kb + bko + h; b[h] = (_Float16)Wh[(k * H + K) * DH + col]; }
        acc = __builtin_amdgcn_wmma_f32_16x16x32_f16(false, a, false, b, (short)0, acc, false, false);
    }
    int rb = (lane < 16) ? 0 : 8;
#pragma unroll
    for (int v = 0; v < 8; ++v) {
        int row = tile * 16 + rb + v;
        float o = acc[v];
        Hh[(k * N + row) * DH + col]   = o;
        Hh_h[(k * N + row) * DH + col] = (_Float16)o;
    }
}

// src[k][n] = Hh[k][n].asrc[k]; dst likewise
__global__ void k_srcdst(const float* __restrict__ Hh, const float* __restrict__ asrc,
                         const float* __restrict__ adst, float* __restrict__ srcv,
                         float* __restrict__ dstv) {
    int id = blockIdx.x * 256 + threadIdx.x;   // 16 blocks x 256 = NH*N
    int k = id >> 10, n = id & (N - 1);
    const float* hp = Hh + ((size_t)k * N + n) * DH;
    float s = 0.f, d = 0.f;
    for (int q = 0; q < DH; ++q) { float h = hp[q]; s += h * asrc[k * DH + q]; d += h * adst[k * DH + q]; }
    srcv[k * N + n] = s; dstv[k * N + n] = d;
}

// Issue a cooperative async copy of a 4 KB B panel (32 K x 64 cols, f16)
// into LDS: 128 threads x (16 B at t*16, 16 B at t*16+2048).
// The 24-bit instruction offset is added to BOTH the LDS and global address
// (ISA 10.x async pseudocode), so one address pair serves both halves.
__device__ __forceinline__ void async_copy_panel(_Float16* lds_dst,
                                                 const _Float16* gsrc, int t) {
    unsigned ldsa = (unsigned)(unsigned long long)lds_dst + (unsigned)(t * 16);
    unsigned long long ga = (unsigned long long)gsrc + (unsigned long long)(t * 16);
    asm volatile("global_load_async_to_lds_b128 %0, %1, off\n\t"
                 "global_load_async_to_lds_b128 %0, %1, off offset:2048"
                 :: "v"(ldsa), "v"(ga) : "memory");
}

// One (step, 16-row tile): per head build masked-softmax w tile in LDS,
// WMMA w@Hh (B panels async-staged + double buffered), elu, concat heads,
// WMMA merged@merge_W, LN, masked partial sum.
__global__ void __launch_bounds__(128)
k_step(const int* __restrict__ ts, const unsigned char* __restrict__ adjm,
       const float* __restrict__ srcv, const float* __restrict__ dstv,
       const _Float16* __restrict__ Hh_h, const _Float16* __restrict__ mWh,
       const float* __restrict__ mb, const float* __restrict__ g2,
       const float* __restrict__ b2, float* __restrict__ rep_part) {
    int tile = blockIdx.x, s = blockIdx.y, t = threadIdx.x;   // 128 threads = 4 waves
    __shared__ _Float16 wt[16 * N];        // 32 KB: softmax weights, one head at a time
    __shared__ _Float16 bpan[2][32 * DH];  // 8 KB : double-buffered async B panels
    __shared__ _Float16 mrg[16 * H];       // 8 KB : concat-head rows (f16 A operand)
    __shared__ float    zt[16 * H];        // 16 KB: pre-LN merge output
    __shared__ float    red[128];
    __shared__ float    part[256];
    const int* act = ts + s * N;
    int lane = t & 31, w = t >> 5;
    int arow = lane & 15, aoff = (lane < 16) ? 0 : 8, bko = (lane < 16) ? 0 : 16;
    int rb = (lane < 16) ? 0 : 8;
    part[t] = 0.f; part[t + 128] = 0.f;

    for (int k = 0; k < NH; ++k) {
        // ---- masked row softmax into wt (mirrors the reference exactly) ----
        for (int r = 0; r < 16; ++r) {
            int i = tile * 16 + r;
            bool acti = act[i] > 0;
            float di = dstv[k * N + i];
            float lmax = -3e30f;
            for (int j = t; j < N; j += 128) {
                bool actj = act[j] > 0;
                bool m = acti && actj && adjm[i * N + j];
                float lr = di + srcv[k * N + j];
                lr = lr > 0.f ? lr : 0.2f * lr;                 // leaky_relu(0.2)
                float val = m ? lr : (actj ? 0.f : -1e30f);     // inactive cols excluded
                lmax = fmaxf(lmax, val);
            }
            red[t] = lmax; __syncthreads();
            for (int o = 64; o; o >>= 1) { if (t < o) red[t] = fmaxf(red[t], red[t + o]); __syncthreads(); }
            float mx = red[0]; __syncthreads();
            float lsum = 0.f;
            for (int j = t; j < N; j += 128) {
                bool actj = act[j] > 0;
                bool m = acti && actj && adjm[i * N + j];
                float lr = di + srcv[k * N + j];
                lr = lr > 0.f ? lr : 0.2f * lr;
                float ex = m ? __expf(lr - mx) : 0.f;
                wt[r * N + j] = (_Float16)ex;
                lsum += ex;
            }
            red[t] = lsum; __syncthreads();
            for (int o = 64; o; o >>= 1) { if (t < o) red[t] += red[t + o]; __syncthreads(); }
            float inv = 1.f / (red[0] + 1e-8f); __syncthreads();
            for (int j = t; j < N; j += 128)
                wt[r * N + j] = (_Float16)((float)wt[r * N + j] * inv);
        }
        __syncthreads();

        // ---- out[k] tile = elu(w @ Hh[k]); B panels async-staged to LDS ----
        const _Float16* B = Hh_h + (size_t)k * N * DH;
        async_copy_panel(&bpan[0][0], B, t);               // prologue: chunk 0
        int col = w * 16 + (lane & 15);
        v8f acc = {};
        for (int kk = 0; kk < 32; ++kk) {
            int cur = kk & 1;
            asm volatile("s_wait_asynccnt 0" ::: "memory"); // panel[cur] landed (this wave)
            __syncthreads();                                // ... and all waves' portions
            if (kk + 1 < 32)                                // overlap DMA with WMMA chain
                async_copy_panel(&bpan[cur ^ 1][0], B + (kk + 1) * 32 * DH, t);
            int kb = kk * 32;
            v16h a, b;
#pragma unroll
            for (int h = 0; h < 16; ++h) { int K = kb + bko + h; a[h] = wt[arow * N + K]; }
#pragma unroll
            for (int h = 0; h < 16; ++h) { b[h] = bpan[cur][(bko + h) * DH + col]; }
            acc = __builtin_amdgcn_wmma_f32_16x16x32_f16(false, a, false, b, (short)0, acc, false, false);
        }
#pragma unroll
        for (int v = 0; v < 8; ++v) {
            float o = acc[v];
            o = o > 0.f ? o : (__expf(o) - 1.f);                // elu
            mrg[(rb + v) * H + k * DH + col] = (_Float16)o;     // concat heads
        }
        __syncthreads();                                        // wt reused next head
    }

    // ---- z = merged @ merge_W + merge_b via WMMA (B pre-converted to f16) ----
    for (int nt = 0; nt < 4; ++nt) {
        int col = w * 64 + nt * 16 + (lane & 15);
        v8f acc = {};
        for (int kk = 0; kk < 8; ++kk) {
            int kb = kk * 32;
            v16h a, b;
#pragma unroll
            for (int h = 0; h < 16; ++h) { int K = kb + bko + h; a[h] = mrg[arow * H + K]; }
#pragma unroll
            for (int h = 0; h < 16; ++h) { int K = kb + bko + h; b[h] = mWh[K * H + col]; }
            acc = __builtin_amdgcn_wmma_f32_16x16x32_f16(false, a, false, b, (short)0, acc, false, false);
        }
#pragma unroll
        for (int v = 0; v < 8; ++v) zt[(rb + v) * H + col] = acc[v] + mb[col];
    }
    __syncthreads();

    // ---- layernorm rows + masked partial mean ----
    for (int r = 0; r < 16; ++r) {
        float x0 = zt[r * H + t], x1 = zt[r * H + t + 128];
        red[t] = x0 + x1; __syncthreads();
        for (int o = 64; o; o >>= 1) { if (t < o) red[t] += red[t + o]; __syncthreads(); }
        float mu = red[0] * (1.f / H); __syncthreads();
        float d0 = x0 - mu, d1 = x1 - mu;
        red[t] = d0 * d0 + d1 * d1; __syncthreads();
        for (int o = 64; o; o >>= 1) { if (t < o) red[t] += red[t + o]; __syncthreads(); }
        float rs = rsqrtf(red[0] * (1.f / H) + 1e-5f); __syncthreads();
        if (act[tile * 16 + r] > 0) {
            part[t]       += d0 * rs * g2[t] + b2[t];
            part[t + 128] += d1 * rs * g2[t + 128] + b2[t + 128];
        }
    }
    __syncthreads();
    size_t o = ((size_t)s * 64 + tile) * H;
    rep_part[o + t] = part[t]; rep_part[o + t + 128] = part[t + 128];
}

// Sequential tail: rep reduction, prev-propagation, GRU scan. One block.
__global__ void k_final(const int* __restrict__ ts, const float* __restrict__ rep_part,
                        const float* __restrict__ no_code, const float* __restrict__ Wih,
                        const float* __restrict__ Whh, const float* __restrict__ bih,
                        const float* __restrict__ bhh, float* __restrict__ out) {
    int t = threadIdx.x;                       // 256 threads = H
    __shared__ float prevv[256], evs[256], hv[256];
    __shared__ int   redi[256];
    prevv[t] = no_code[t]; hv[t] = 0.f;
    __syncthreads();
    for (int s = 0; s < S; ++s) {
        int c = 0;
        for (int n = t; n < N; n += 256) c += (ts[s * N + n] > 0) ? 1 : 0;
        redi[t] = c; __syncthreads();
        for (int o = 128; o; o >>= 1) { if (t < o) redi[t] += redi[t + o]; __syncthreads(); }
        int cnt = redi[0]; __syncthreads();
        float rsum = 0.f;
        for (int tl = 0; tl < 64; ++tl) rsum += rep_part[((size_t)s * 64 + tl) * H + t];
        float rep = rsum / (float)(cnt > 0 ? cnt : 1);
        float ev = cnt > 0 ? rep : prevv[t];
        evs[t] = ev; prevv[t] = ev;
        __syncthreads();
        // GRU step: thread t owns output index t of each of the 3 gates
        float gi0 = bih[t], gi1 = bih[H + t], gi2 = bih[2 * H + t];
        float gh0 = bhh[t], gh1 = bhh[H + t], gh2 = bhh[2 * H + t];
        for (int h = 0; h < H; ++h) {
            float x = evs[h], hh = hv[h];
            gi0 += Wih[t * H + h] * x;
            gi1 += Wih[(H + t) * H + h] * x;
            gi2 += Wih[(2 * H + t) * H + h] * x;
            gh0 += Whh[t * H + h] * hh;
            gh1 += Whh[(H + t) * H + h] * hh;
            gh2 += Whh[(2 * H + t) * H + h] * hh;
        }
        float r_ = 1.f / (1.f + __expf(-(gi0 + gh0)));
        float z_ = 1.f / (1.f + __expf(-(gi1 + gh1)));
        float n_ = tanhf(gi2 + r_ * gh2);
        float hn = (1.f - z_) * n_ + z_ * hv[t];
        __syncthreads();
        hv[t] = hn;
        __syncthreads();
    }
    out[t] = hv[t];
}

extern "C" void kernel_launch(void* const* d_in, const int* in_sizes, int n_in,
                              void* d_out, int out_size, void* d_ws, size_t ws_size,
                              hipStream_t stream) {
    const int*   ts     = (const int*)  d_in[0];
    const float* adj    = (const float*)d_in[1];
    const float* emb    = (const float*)d_in[2];
    const float* g1     = (const float*)d_in[3];
    const float* b1     = (const float*)d_in[4];
    const float* Wh     = (const float*)d_in[5];
    const float* asrc   = (const float*)d_in[6];
    const float* adst   = (const float*)d_in[7];
    const float* mW     = (const float*)d_in[8];
    const float* mb     = (const float*)d_in[9];
    const float* g2     = (const float*)d_in[10];
    const float* b2     = (const float*)d_in[11];
    const float* gbW1   = (const float*)d_in[12];
    const float* gbb1   = (const float*)d_in[13];
    const float* gbw2   = (const float*)d_in[14];
    const float* gbb2   = (const float*)d_in[15];
    const float* alpha  = (const float*)d_in[16];
    const float* pw     = (const float*)d_in[17];
    const float* pb     = (const float*)d_in[18];
    const float* nocode = (const float*)d_in[19];
    const float* Wih    = (const float*)d_in[20];
    const float* Whh    = (const float*)d_in[21];
    const float* bih    = (const float*)d_in[22];
    const float* bhh    = (const float*)d_in[23];
    (void)in_sizes; (void)n_in; (void)out_size;

    // workspace layout (~6.0 MB total)
    float* ws   = (float*)d_ws;
    float* e    = ws;                         // 262144 f32
    float* ua   = e + N * H;                  // 256
    float* ub   = ua + H;                     // 256
    float* cc   = ub + H;                     // 4 (pad)
    float* av   = cc + 4;                     // 1024
    float* bv   = av + N;                     // 1024
    float* Hh   = bv + N;                     // 262144
    float* srcv = Hh + NH * N * DH;           // 4096
    float* dstv = srcv + NH * N;              // 4096
    float* rep_part = dstv + NH * N;          // S*64*H = 524288
    _Float16* Hh_h = (_Float16*)(rep_part + (size_t)S * 64 * H);   // 262144 halves
    _Float16* mWh  = Hh_h + NH * N * DH;                           // 65536 halves
    unsigned char* adjm = (unsigned char*)(mWh + H * H);           // N*N bytes
    (void)ws_size;

    k_uabc  <<<1,            256, 0, stream>>>(gbW1, gbb1, gbw2, gbb2, ua, ub, cc);
    k_ln1   <<<N,            256, 0, stream>>>(emb, g1, b1, ua, ub, e, av, bv);
    k_adj   <<<N,            256, 0, stream>>>(adj, av, bv, cc, alpha, pw, pb, adjm);
    k_cvtw  <<<H * H / 256,  256, 0, stream>>>(mW, mWh);
    k_hh    <<<dim3(64, NH), 128, 0, stream>>>(e, Wh, Hh, Hh_h);
    k_srcdst<<<16,           256, 0, stream>>>(Hh, asrc, adst, srcv, dstv);
    k_step  <<<dim3(64, S),  128, 0, stream>>>(ts, adjm, srcv, dstv, Hh_h, mWh,
                                               mb, g2, b2, rep_part);
    k_final <<<1,            256, 0, stream>>>(ts, rep_part, nocode, Wih, Whh, bih, bhh,
                                               (float*)d_out);
}
